// EfficientSelfAttention_81295140979160
// MI455X (gfx1250) — matrix operand order repaired and checked by reading, not compiled, not run on hardware
//
#include <hip/hip_runtime.h>

#define BATCH 4
#define SEQ   4096
#define DH    1024
#define NHEAD 16
#define DHEAD 64
#define MTOT  (BATCH*SEQ)   /* 16384 */
#define K3    3072
#define QVS   2048          /* packed qkv output row stride (q-part | v-part) */

static constexpr float S_SCALE = 0.5946035575013605f; /* 8^(-1/4) */

typedef __attribute__((ext_vector_type(16))) __bf16 v16bf;
typedef __attribute__((ext_vector_type(8)))  float  v8f;

union FragBF { v16bf v; uint4 q[2]; };

__device__ __forceinline__ unsigned short f2bf(float x) {
  union { float f; unsigned u; } c; c.f = x;
  unsigned u = c.u;
  u += 0x7FFFu + ((u >> 16) & 1u);   /* round-to-nearest-even */
  return (unsigned short)(u >> 16);
}

__device__ __forceinline__ unsigned lds_off(const void* p) {
  /* flat pointers into the LDS aperture carry the wave-relative LDS byte
     address in their low 32 bits (CDNA5 aperture mapping) */
  return (unsigned)(uintptr_t)p;
}

__device__ __forceinline__ v8f wmma_bf16(v16bf a, v16bf b, v8f c) {
  return __builtin_amdgcn_wmma_f32_16x16x32_bf16(false, a, false, b, (short)0, c,
                                                 false, false);
}

/* A fragment, LDS tile row-major [M][K]:
   lane<16: M=lane, K={0..7,16..23}; lane>=16: M=lane-16, K={8..15,24..31}. */
__device__ __forceinline__ v16bf frag_a(const unsigned short* base, int stride,
                                        int row, int lo) {
  FragBF f;
  const unsigned short* p = base + row * stride;
  f.q[0] = *(const uint4*)(p + lo);
  f.q[1] = *(const uint4*)(p + lo + 16);
  return f.v;
}
/* B fragment, LDS tile transposed [N][K]:
   lane<16: N=lane, K={0..7,16..23}; lane>=16: N=lane-16, K={8..15,24..31}. */
__device__ __forceinline__ v16bf frag_b(const unsigned short* base, int stride,
                                        int col, int hi) {
  FragBF f;
  const unsigned short* p = base + col * stride;
  f.q[0] = *(const uint4*)(p + hi / 2);
  f.q[1] = *(const uint4*)(p + hi / 2 + 16);
  return f.v;
}

/* issue one 128x32 bf16 tile pair (A,B) via async HBM/L2 -> LDS copies */
__device__ __forceinline__ void issue_tiles(
    const unsigned short* A, const unsigned short* B,
    const unsigned* ldsA, const unsigned* ldsB,
    const unsigned* gA, const unsigned* gB,
    unsigned kbyte, unsigned bufoff)
{
  #pragma unroll
  for (int i = 0; i < 2; ++i) {
    asm volatile("global_load_async_to_lds_b128 %0, %1, %2"
                 :: "v"(ldsA[i] + bufoff), "v"(gA[i] + kbyte), "s"(A) : "memory");
    asm volatile("global_load_async_to_lds_b128 %0, %1, %2"
                 :: "v"(ldsB[i] + bufoff), "v"(gB[i] + kbyte), "s"(B) : "memory");
  }
}

/* ============ packing kernels: one-shot fp32 -> bf16 (L2-resident GEMM feeds) */
__global__ __launch_bounds__(256) void pack_x(
    const float* __restrict__ q, const float* __restrict__ k,
    const float* __restrict__ v, unsigned short* __restrict__ X)
{
  int m = blockIdx.x;
  int s = blockIdx.y;
  const float* src = (s == 0) ? q : (s == 1) ? k : v;
  int c4 = threadIdx.x * 4;
  float4 x = *(const float4*)(src + (size_t)m * DH + c4);
  union { unsigned short us[4]; uint2 u2; } o;
  o.us[0] = f2bf(x.x); o.us[1] = f2bf(x.y); o.us[2] = f2bf(x.z); o.us[3] = f2bf(x.w);
  *(uint2*)(X + (size_t)m * K3 + s * DH + c4) = o.u2;
}

__global__ __launch_bounds__(256) void pack_wqkv(const float* __restrict__ W,
                                                 unsigned short* __restrict__ Wp)
{
  int r  = blockIdx.x;                     /* packed row 0..2047 */
  int ng = (r < 1024) ? r : r + 1024;      /* skip dead K-projection rows */
  const float* src = W + (size_t)ng * K3;
  unsigned short* dst = Wp + (size_t)r * K3;
  #pragma unroll
  for (int i = 0; i < 3; ++i) {
    int c4 = threadIdx.x * 4 + i * 1024;
    float4 x = *(const float4*)(src + c4);
    union { unsigned short us[4]; uint2 u2; } o;
    o.us[0] = f2bf(x.x); o.us[1] = f2bf(x.y); o.us[2] = f2bf(x.z); o.us[3] = f2bf(x.w);
    *(uint2*)(dst + c4) = o.u2;
  }
}

__global__ __launch_bounds__(256) void pack_wout(const float* __restrict__ W,
                                                 unsigned short* __restrict__ Wo)
{
  int r = blockIdx.x;
  int c4 = threadIdx.x * 4;
  float4 x = *(const float4*)(W + (size_t)r * DH + c4);
  union { unsigned short us[4]; uint2 u2; } o;
  o.us[0] = f2bf(x.x); o.us[1] = f2bf(x.y); o.us[2] = f2bf(x.z); o.us[3] = f2bf(x.w);
  *(uint2*)(Wo + (size_t)r * DH + c4) = o.u2;
}

/* ============ QKV GEMM: double-buffered async-LDS staging, WMMA core ========
   QV[m][c] = X[m] . Wp[c] + b[c<1024 ? c : c+1024]  (c in [0,2048)) */
#define BKP 40
#define TILEB (128 * BKP * 2)   /* bytes per LDS tile buffer */

__global__ __launch_bounds__(256) void qkv_gemm(
    const unsigned short* __restrict__ X, const unsigned short* __restrict__ Wp,
    const float* __restrict__ bias, float* __restrict__ QV)
{
  __shared__ unsigned short As[2][128 * BKP];
  __shared__ unsigned short Bs[2][128 * BKP];
  const int tid  = threadIdx.x;
  const int lane = tid & 31;
  const int wave = tid >> 5;
  const int mBase = blockIdx.y * 128;
  const int cb    = blockIdx.x * 128;
  const int wm = (wave & 1) * 64;
  const int wn = (wave >> 1) * 32;
  const int alo = (lane < 16) ? 0 : 8;
  const int bhi = (lane < 16) ? 0 : 16;
  const int nl  = lane & 15;
  v8f acc[4][2] = {};

  /* async chunk addressing: chunk c = tid + 256*i ; 512 16B chunks per tile */
  unsigned ldsA[2], ldsB[2], gA[2], gB[2];
  #pragma unroll
  for (int i = 0; i < 2; ++i) {
    int c   = tid + 256 * i;
    int row = c >> 2;
    int ke  = (c & 3) * 8;
    ldsA[i] = lds_off(As[0]) + (unsigned)(row * BKP + ke) * 2;
    ldsB[i] = lds_off(Bs[0]) + (unsigned)(row * BKP + ke) * 2;
    gA[i]   = (unsigned)(((mBase + row) * K3 + ke) * 2);
    gB[i]   = (unsigned)(((cb    + row) * K3 + ke) * 2);
  }

  const int NT = K3 / 32;
  issue_tiles(X, Wp, ldsA, ldsB, gA, gB, 0u, 0u);
  #pragma clang loop unroll(disable)
  for (int t = 0; t < NT; ++t) {
    if (t + 1 < NT) {
      issue_tiles(X, Wp, ldsA, ldsB, gA, gB,
                  (unsigned)(t + 1) * 64u, (unsigned)((t + 1) & 1) * TILEB);
      asm volatile("s_wait_asynccnt 4" ::: "memory"); /* prev batch landed */
    } else {
      asm volatile("s_wait_asynccnt 0" ::: "memory");
    }
    __syncthreads();
    const unsigned short* Ab = As[t & 1];
    const unsigned short* Bb = Bs[t & 1];
    v16bf b0 = frag_b(Bb, BKP, wn + nl,      bhi);
    v16bf b1 = frag_b(Bb, BKP, wn + 16 + nl, bhi);
    #pragma unroll
    for (int mt = 0; mt < 4; ++mt) {
      v16bf a = frag_a(Ab, BKP, wm + 16 * mt + nl, alo);
      acc[mt][0] = wmma_bf16(a, b0, acc[mt][0]);
      acc[mt][1] = wmma_bf16(a, b1, acc[mt][1]);
    }
    __syncthreads();
  }

  const int hi8 = (lane >> 4) * 8;
  #pragma unroll
  for (int mt = 0; mt < 4; ++mt)
    #pragma unroll
    for (int nt = 0; nt < 2; ++nt) {
      int m0 = mBase + wm + 16 * mt + hi8;
      int c0 = cb + wn + 16 * nt + nl;
      int ng = (c0 < 1024) ? c0 : c0 + 1024;
      float bv = bias[ng];
      #pragma unroll
      for (int r = 0; r < 8; ++r)
        QV[(size_t)(m0 + r) * QVS + c0] = acc[mt][nt][r] + bv;
    }
}

/* ============ per-head row softmax: qs = softmax(qh,-1)*S (in-place q-part) = */
__global__ __launch_bounds__(256) void head_softmax(float* __restrict__ QV)
{
  int gw   = (blockIdx.x * blockDim.x + threadIdx.x) >> 5;
  int lane = threadIdx.x & 31;
  int m = gw >> 4;
  int h = gw & 15;
  float* row = QV + (size_t)m * QVS + h * DHEAD;
  float x0 = row[lane], x1 = row[lane + 32];
  float mx = fmaxf(x0, x1);
  #pragma unroll
  for (int off = 16; off; off >>= 1) mx = fmaxf(mx, __shfl_xor(mx, off, 32));
  float e0 = __expf(x0 - mx), e1 = __expf(x1 - mx);
  float s = e0 + e1;
  #pragma unroll
  for (int off = 16; off; off >>= 1) s += __shfl_xor(s, off, 32);
  float inv = S_SCALE / s;
  row[lane] = e0 * inv;
  row[lane + 32] = e1 * inv;
}

/* ============ column stats over n for the ks_ softmax ======================= */
__global__ __launch_bounds__(256) void col_stats(const float* __restrict__ QV,
                                                 float* __restrict__ cmax,
                                                 float* __restrict__ cscale)
{
  int bh = blockIdx.x;
  int b = bh >> 4, h = bh & 15;
  int d  = threadIdx.x & 63;
  int st = threadIdx.x >> 6;
  const float* base = QV + (size_t)b * SEQ * QVS + h * DHEAD + d;
  float m = -3.0e38f, s = 0.f;
  for (int n = st; n < SEQ; n += 4) {
    float x  = base[(size_t)n * QVS];
    float nm = fmaxf(m, x);
    s = s * __expf(m - nm) + __expf(x - nm);
    m = nm;
  }
  __shared__ float sm[4][64], ss[4][64];
  sm[st][d] = m; ss[st][d] = s;
  __syncthreads();
  if (st == 0) {
    float M = sm[0][d], Sv = ss[0][d];
    #pragma unroll
    for (int i = 1; i < 4; ++i) {
      float m2 = sm[i][d], s2 = ss[i][d];
      float nm = fmaxf(M, m2);
      Sv = Sv * __expf(M - nm) + s2 * __expf(m2 - nm);
      M = nm;
    }
    cmax[bh * 64 + d]   = M;
    cscale[bh * 64 + d] = S_SCALE / Sv;
  }
}

/* ============ context[bh] = ks_^T (64 x n) . vh (n x 64), WMMA over n ======= */
#define CKP 40
__global__ __launch_bounds__(256) void context_kernel(
    const float* __restrict__ QV,
    const float* __restrict__ cmax, const float* __restrict__ cscale,
    float* __restrict__ CTX)
{
  __shared__ unsigned short KT[64 * CKP];  /* [d][n] */
  __shared__ unsigned short VT[64 * CKP];  /* [e][n] */
  int bh = blockIdx.x;
  int b = bh >> 4, h = bh & 15;
  int tid = threadIdx.x, lane = tid & 31, wave = tid >> 5;
  int d   = tid & 63;
  int nl0 = tid >> 6;
  const float* qbase = QV + (size_t)b * SEQ * QVS + h * DHEAD;
  const float* vbase = qbase + 1024;       /* v-part lives in cols [1024,2048) */
  float cm = cmax[bh * 64 + d], cs = cscale[bh * 64 + d];
  const int alo = (lane < 16) ? 0 : 8;
  const int bhi = (lane < 16) ? 0 : 16;
  const int nl  = lane & 15;
  int dt  = wave >> 1;
  int et0 = (wave & 1) * 2;
  v8f acc[2] = {};

  #pragma clang loop unroll(disable)
  for (int n0 = 0; n0 < SEQ; n0 += 32) {
    #pragma unroll
    for (int i = 0; i < 8; ++i) {
      int nn = nl0 + 4 * i;
      float qv = qbase[(size_t)(n0 + nn) * QVS + d];
      KT[d * CKP + nn] = f2bf(__expf(qv - cm) * cs);
      float vv = vbase[(size_t)(n0 + nn) * QVS + d];
      VT[d * CKP + nn] = f2bf(vv);
    }
    __syncthreads();
    v16bf a  = frag_a(KT, CKP, 16 * dt + nl, alo);
    v16bf b0 = frag_b(VT, CKP, 16 * et0 + nl, bhi);
    v16bf b1 = frag_b(VT, CKP, 16 * (et0 + 1) + nl, bhi);
    acc[0] = wmma_bf16(a, b0, acc[0]);
    acc[1] = wmma_bf16(a, b1, acc[1]);
    __syncthreads();
  }

  int hi8 = (lane >> 4) * 8;
  float* cb = CTX + (size_t)bh * 64 * 64;
  #pragma unroll
  for (int t = 0; t < 2; ++t) {
    int eg = 16 * (et0 + t) + nl;
    #pragma unroll
    for (int r = 0; r < 8; ++r)
      cb[(16 * dt + hi8 + r) * 64 + eg] = acc[t][r];
  }
}

/* ============ att = qs (n x 64) . context (64 x 64) -> bf16 ================= */
#define EKP 72
__global__ __launch_bounds__(256) void att_kernel(
    const float* __restrict__ QV, const float* __restrict__ CTX,
    unsigned short* __restrict__ ATT16)
{
  __shared__ unsigned short As[128 * EKP]; /* [m][d] */
  __shared__ unsigned short Bs[64 * EKP];  /* [e][d] */
  int bh = blockIdx.x;
  int nt = blockIdx.y;
  int b = bh >> 4, h = bh & 15;
  int tid = threadIdx.x, lane = tid & 31, wave = tid >> 5;
  const float* qbase = QV + (size_t)b * SEQ * QVS + h * DHEAD
                          + (size_t)nt * 128 * QVS;
  {
    int c4 = (tid & 15) * 4;
    int r0 = tid >> 4;
    #pragma unroll
    for (int i = 0; i < 8; ++i) {
      int r = r0 + 16 * i;
      float4 x = *(const float4*)(qbase + (size_t)r * QVS + c4);
      unsigned short* d = As + r * EKP + c4;
      d[0] = f2bf(x.x); d[1] = f2bf(x.y); d[2] = f2bf(x.z); d[3] = f2bf(x.w);
    }
    const float* cb = CTX + (size_t)bh * 64 * 64;
    int e = tid & 63, d0 = tid >> 6;
    #pragma unroll
    for (int i = 0; i < 16; ++i) {
      int dd = d0 + 4 * i;
      Bs[e * EKP + dd] = f2bf(cb[dd * 64 + e]);
    }
  }
  __syncthreads();
  const int alo = (lane < 16) ? 0 : 8;
  const int bhi = (lane < 16) ? 0 : 16;
  const int nl  = lane & 15;
  v8f acc[4] = {};
  #pragma unroll
  for (int ks = 0; ks < 64; ks += 32) {
    v16bf a = frag_a(As + ks, EKP, 16 * wave + nl, alo);
    #pragma unroll
    for (int et = 0; et < 4; ++et) {
      v16bf bb = frag_b(Bs + ks, EKP, 16 * et + nl, bhi);
      acc[et] = wmma_bf16(a, bb, acc[et]);
    }
  }
  int hi8 = (lane >> 4) * 8;
  unsigned short* ob = ATT16 + ((size_t)b * SEQ + (size_t)nt * 128) * DH + h * DHEAD;
  #pragma unroll
  for (int et = 0; et < 4; ++et)
    #pragma unroll
    for (int r = 0; r < 8; ++r)
      ob[(size_t)(16 * wave + hi8 + r) * DH + 16 * et + nl] = f2bf(acc[et][r]);
}

/* ============ out = ATT16 . Wo16^T + b_out, double-buffered async, WMMA ===== */
__global__ __launch_bounds__(256) void out_gemm(
    const unsigned short* __restrict__ A16, const unsigned short* __restrict__ W16,
    const float* __restrict__ bias, float* __restrict__ OUT)
{
  __shared__ unsigned short As[2][128 * BKP];
  __shared__ unsigned short Bs[2][128 * BKP];
  const int tid  = threadIdx.x;
  const int lane = tid & 31;
  const int wave = tid >> 5;
  const int mBase = blockIdx.y * 128;
  const int nBase = blockIdx.x * 128;
  const int wm = (wave & 1) * 64;
  const int wn = (wave >> 1) * 32;
  const int alo = (lane < 16) ? 0 : 8;
  const int bhi = (lane < 16) ? 0 : 16;
  const int nl  = lane & 15;
  v8f acc[4][2] = {};

  unsigned ldsA[2], ldsB[2], gA[2], gB[2];
  #pragma unroll
  for (int i = 0; i < 2; ++i) {
    int c   = tid + 256 * i;
    int row = c >> 2;
    int ke  = (c & 3) * 8;
    ldsA[i] = lds_off(As[0]) + (unsigned)(row * BKP + ke) * 2;
    ldsB[i] = lds_off(Bs[0]) + (unsigned)(row * BKP + ke) * 2;
    gA[i]   = (unsigned)(((mBase + row) * DH + ke) * 2);
    gB[i]   = (unsigned)(((nBase + row) * DH + ke) * 2);
  }

  const int NT = DH / 32;
  issue_tiles(A16, W16, ldsA, ldsB, gA, gB, 0u, 0u);
  #pragma clang loop unroll(disable)
  for (int t = 0; t < NT; ++t) {
    if (t + 1 < NT) {
      issue_tiles(A16, W16, ldsA, ldsB, gA, gB,
                  (unsigned)(t + 1) * 64u, (unsigned)((t + 1) & 1) * TILEB);
      asm volatile("s_wait_asynccnt 4" ::: "memory");
    } else {
      asm volatile("s_wait_asynccnt 0" ::: "memory");
    }
    __syncthreads();
    const unsigned short* Ab = As[t & 1];
    const unsigned short* Bb = Bs[t & 1];
    v16bf b0 = frag_b(Bb, BKP, wn + nl,      bhi);
    v16bf b1 = frag_b(Bb, BKP, wn + 16 + nl, bhi);
    #pragma unroll
    for (int mt = 0; mt < 4; ++mt) {
      v16bf a = frag_a(Ab, BKP, wm + 16 * mt + nl, alo);
      acc[mt][0] = wmma_bf16(a, b0, acc[mt][0]);
      acc[mt][1] = wmma_bf16(a, b1, acc[mt][1]);
    }
    __syncthreads();
  }

  int hi8 = (lane >> 4) * 8;
  #pragma unroll
  for (int mt = 0; mt < 4; ++mt)
    #pragma unroll
    for (int nt = 0; nt < 2; ++nt) {
      int m0 = mBase + wm + 16 * mt + hi8;
      int n0 = nBase + wn + 16 * nt + nl;
      float bv = bias[n0];
      #pragma unroll
      for (int r = 0; r < 8; ++r)
        OUT[(size_t)(m0 + r) * DH + n0] = acc[mt][nt][r] + bv;
    }
}

extern "C" void kernel_launch(void* const* d_in, const int* in_sizes, int n_in,
                              void* d_out, int out_size, void* d_ws, size_t ws_size,
                              hipStream_t stream)
{
  (void)in_sizes; (void)n_in; (void)out_size; (void)ws_size;
  const float* q    = (const float*)d_in[0];
  const float* k    = (const float*)d_in[1];
  const float* v    = (const float*)d_in[2];
  const float* Wqkv = (const float*)d_in[3];
  const float* bqkv = (const float*)d_in[4];
  const float* Wout = (const float*)d_in[5];
  const float* bout = (const float*)d_in[6];

  char* w = (char*)d_ws;
  float* QV = (float*)w;                 w += (size_t)MTOT * QVS * 4;   /* 128MB */
  unsigned short* X16 = (unsigned short*)w; w += (size_t)MTOT * K3 * 2; /*  96MB */
  unsigned short* Wp  = (unsigned short*)w; w += (size_t)2048 * K3 * 2; /*  12MB */
  unsigned short* Wo  = (unsigned short*)w; w += (size_t)DH * DH * 2;   /*   2MB */
  unsigned short* ATT = (unsigned short*)w; w += (size_t)MTOT * DH * 2; /*  32MB */
  float* CTX  = (float*)w;               w += (size_t)64 * 64 * 64 * 4;
  float* CMAX = (float*)w;               w += 4096 * 4;
  float* CSCL = (float*)w;

  pack_x        <<<dim3(MTOT, 3), 256, 0, stream>>>(q, k, v, X16);
  pack_wqkv     <<<2048, 256, 0, stream>>>(Wqkv, Wp);
  pack_wout     <<<1024, 256, 0, stream>>>(Wout, Wo);
  qkv_gemm      <<<dim3(16, 128), 256, 0, stream>>>(X16, Wp, bqkv, QV);
  head_softmax  <<<(MTOT * NHEAD) / 8, 256, 0, stream>>>(QV);
  col_stats     <<<BATCH * NHEAD, 256, 0, stream>>>(QV, CMAX, CSCL);
  context_kernel<<<BATCH * NHEAD, 256, 0, stream>>>(QV, CMAX, CSCL, CTX);
  att_kernel    <<<dim3(BATCH * NHEAD, SEQ / 128), 256, 0, stream>>>(QV, CTX, ATT);
  out_gemm      <<<dim3(8, 128), 256, 0, stream>>>(ATT, Wo, bout, (float*)d_out);
}
